// JointAttention_70093866271241
// MI455X (gfx1250) — matrix operand order repaired and checked by reading, not compiled, not run on hardware
//
#include <hip/hip_runtime.h>

typedef __bf16 bf16;
typedef __attribute__((ext_vector_type(16))) bf16  v16bf;
typedef __attribute__((ext_vector_type(8)))  bf16  v8bf;
typedef __attribute__((ext_vector_type(8)))  float v8f;

#define DEV __device__ __forceinline__

// Problem constants
#define BB   2
#define HH   16
#define DD   64
#define NN0  1536
#define NN1  512
#define NNT  2048      // packed sequence
#define DI   1024      // H*D
#define DIM0 1024
#define DIM1 512
#define K3   3072      // 3*DI

DEV v16bf cat8(v8bf lo, v8bf hi) {
  v16bf r;
#pragma unroll
  for (int i = 0; i < 8; ++i) { r[i] = lo[i]; r[i + 8] = hi[i]; }
  return r;
}

DEV v8f wmma_bf16(v16bf a, v16bf b, v8f c) {
  // D = A(16x32 bf16) x B(32x16 bf16) + C(16x16 f32)
  return __builtin_amdgcn_wmma_f32_16x16x32_bf16(false, a, false, b, (short)0, c,
                                                 false, false);
}

DEV float fast_tanh(float x) {
#if __has_builtin(__builtin_amdgcn_tanhf)
  return __builtin_amdgcn_tanhf(x);   // V_TANH_F32
#else
  return tanhf(x);
#endif
}

// exp(50*t) folded into a single exp2: 50*log2(e) = 72.13475204444817
DEV float exp50(float t) {
#if __has_builtin(__builtin_amdgcn_exp2f)
  return __builtin_amdgcn_exp2f(72.134752044f * t);
#else
  return __expf(50.0f * t);
#endif
}

// ---------------------------------------------------------------------------
// f32 -> bf16 elementwise convert
// ---------------------------------------------------------------------------
__global__ __launch_bounds__(256)
void cast_f32_bf16(const float* __restrict__ src, bf16* __restrict__ dst,
                   long long n) {
  long long i = (long long)blockIdx.x * blockDim.x + threadIdx.x;
  long long stride = (long long)gridDim.x * blockDim.x;
  for (; i < n; i += stride) dst[i] = (bf16)src[i];
}

// f32 [rows][cols] -> bf16 [cols][rows]   (weight transpose for B fragments)
__global__ __launch_bounds__(256)
void transpose_f32_bf16(const float* __restrict__ src, bf16* __restrict__ dst,
                        int rows, int cols) {
  long long n = (long long)rows * cols;
  long long i = (long long)blockIdx.x * blockDim.x + threadIdx.x;
  long long stride = (long long)gridDim.x * blockDim.x;
  for (; i < n; i += stride) {
    int r = (int)(i / cols);
    int c = (int)(i % cols);
    dst[(long long)c * rows + r] = (bf16)src[i];
  }
}

// ---------------------------------------------------------------------------
// Generic GEMM: C[M,N] = A[M,K] * B[K,N], with B stored transposed Bt[N][K].
// One wave computes a 16x64 tile (1 A-frag reused by 4 B-frags per K-step).
// grid = (N/64, M/16, batch)
// ---------------------------------------------------------------------------
template <bool OUT_F32>
__global__ __launch_bounds__(32)
void gemm16x64(const bf16* __restrict__ A, const bf16* __restrict__ Bt,
               void* __restrict__ C, int M, int N, int K,
               long long sA, long long sC) {
  const int lane = threadIdx.x & 31;
  const int half = lane >> 4;
  const int l16  = lane & 15;
  const int tN = blockIdx.x * 64;
  const int tM = blockIdx.y * 16;
  const bf16* Ab = A + (long long)blockIdx.z * sA;

  v8f acc0 = {0,0,0,0,0,0,0,0};
  v8f acc1 = acc0, acc2 = acc0, acc3 = acc0;

  // A frag: lanes 0-15 row=l16 K{0..15}; lanes 16-31 K{16..31}
  const bf16* ar = Ab + (long long)(tM + l16) * K + (half << 4);
  // B frag: lane holds column tN+l16, 16 contiguous K values (halves split 0..15/16..31)
  const bf16* b0 = Bt + (long long)(tN +  0 + l16) * K + (half << 4);
  const bf16* b1 = Bt + (long long)(tN + 16 + l16) * K + (half << 4);
  const bf16* b2 = Bt + (long long)(tN + 32 + l16) * K + (half << 4);
  const bf16* b3 = Bt + (long long)(tN + 48 + l16) * K + (half << 4);

  for (int k0 = 0; k0 < K; k0 += 32) {
    v16bf af  = cat8(*(const v8bf*)(ar + k0), *(const v8bf*)(ar + k0 + 8));
    v16bf fb0 = *(const v16bf*)(b0 + k0);
    v16bf fb1 = *(const v16bf*)(b1 + k0);
    v16bf fb2 = *(const v16bf*)(b2 + k0);
    v16bf fb3 = *(const v16bf*)(b3 + k0);
    acc0 = wmma_bf16(af, fb0, acc0);
    acc1 = wmma_bf16(af, fb1, acc1);
    acc2 = wmma_bf16(af, fb2, acc2);
    acc3 = wmma_bf16(af, fb3, acc3);
  }

  long long cb = (long long)blockIdx.z * sC;
#pragma unroll
  for (int v = 0; v < 8; ++v) {
    long long row = tM + v + half * 8;
    long long o = cb + row * N + tN + l16;
    if (OUT_F32) {
      float* Cf = (float*)C;
      Cf[o] = acc0[v]; Cf[o + 16] = acc1[v]; Cf[o + 32] = acc2[v]; Cf[o + 48] = acc3[v];
    } else {
      bf16* Cb = (bf16*)C;
      Cb[o] = (bf16)acc0[v]; Cb[o + 16] = (bf16)acc1[v];
      Cb[o + 32] = (bf16)acc2[v]; Cb[o + 48] = (bf16)acc3[v];
    }
  }
}

// ---------------------------------------------------------------------------
// QK RMSNorm + head split. qkv: bf16 [b][N][3*H*D] (sel-major columns).
// Writes q,k -> bf16 [b][h][n][d]; v -> bf16 transposed [b][h][d][n].
// One wave per (b,n,h); 2 elements per lane.
// ---------------------------------------------------------------------------
__global__ __launch_bounds__(256)
void qknorm_split(const bf16* __restrict__ qkv,
                  const float* __restrict__ qg0, const float* __restrict__ kg0,
                  const float* __restrict__ qg1, const float* __restrict__ kg1,
                  bf16* __restrict__ Qo, bf16* __restrict__ Ko,
                  bf16* __restrict__ Vt) {
  int wid  = blockIdx.x * 8 + (threadIdx.x >> 5);
  int lane = threadIdx.x & 31;
  int h = wid & 15;
  int n = (wid >> 4) & (NNT - 1);
  int b = wid >> 15;

  const bf16* row = qkv + (long long)(b * NNT + n) * K3;
  const float* qg = (n < NN0) ? qg0 : qg1;
  const float* kg = (n < NN0) ? kg0 : kg1;
  int d = lane * 2;
  long long qoff = ((long long)((b * HH + h) * NNT + n)) * DD + d;

  // Q
  float a0 = (float)row[h * DD + d], a1 = (float)row[h * DD + d + 1];
  float ss = a0 * a0 + a1 * a1;
#pragma unroll
  for (int m = 1; m < 32; m <<= 1) ss += __shfl_xor(ss, m, 32);
  float inv = 8.0f / fmaxf(sqrtf(ss), 1e-12f);       // sqrt(D)=8
  Qo[qoff]     = (bf16)(a0 * inv * qg[h * DD + d]);
  Qo[qoff + 1] = (bf16)(a1 * inv * qg[h * DD + d + 1]);

  // K
  float c0 = (float)row[DI + h * DD + d], c1 = (float)row[DI + h * DD + d + 1];
  float s2 = c0 * c0 + c1 * c1;
#pragma unroll
  for (int m = 1; m < 32; m <<= 1) s2 += __shfl_xor(s2, m, 32);
  float inv2 = 8.0f / fmaxf(sqrtf(s2), 1e-12f);
  Ko[qoff]     = (bf16)(c0 * inv2 * kg[h * DD + d]);
  Ko[qoff + 1] = (bf16)(c1 * inv2 * kg[h * DD + d + 1]);

  // V (transposed for contiguous PV B-fragments)
  bf16 v0 = row[2 * DI + h * DD + d];
  bf16 v1 = row[2 * DI + h * DD + d + 1];
  long long vb = (long long)(b * HH + h) * DD * NNT;
  Vt[vb + (long long)d       * NNT + n] = v0;
  Vt[vb + (long long)(d + 1) * NNT + n] = v1;
}

// ---------------------------------------------------------------------------
// Flash attention with tanh soft-clamp. One wave per 32-query tile.
// Scores are bounded to [-50,50] by the clamp, so exp() cannot overflow:
// fixed-offset softmax, no running max / no rescaling. Row sums are kept as
// per-lane partials and reduced once in the epilogue.
// Q,K: bf16 [b][h][n][64]; Vt: bf16 [b][h][64][n]; O: bf16 [b][n][H*64].
// grid = (N/32, H, B)
// ---------------------------------------------------------------------------
__global__ __launch_bounds__(32)
void attn_fwd(const bf16* __restrict__ Q, const bf16* __restrict__ Kt,
              const bf16* __restrict__ Vt, bf16* __restrict__ O) {
  const int lane = threadIdx.x & 31;
  const int half = lane >> 4;
  const int l16  = lane & 15;
  const int qt = blockIdx.x * 32;
  const int h  = blockIdx.y;
  const int b  = blockIdx.z;
  const long long bh = (long long)(b * HH + h);
  const bf16* qp = Q  + bh * NNT * DD;
  const bf16* kp = Kt + bh * NNT * DD;
  const bf16* vp = Vt + bh * (long long)DD * NNT;

  // Q A-fragments: 2 row-tiles x 2 K-slices (d[0..31], d[32..63])
  v16bf aq[2][2];
#pragma unroll
  for (int rt = 0; rt < 2; ++rt) {
    const bf16* qr = qp + (long long)(qt + rt * 16 + l16) * DD + (half << 4);
    aq[rt][0] = cat8(*(const v8bf*)(qr),      *(const v8bf*)(qr + 8));
    aq[rt][1] = cat8(*(const v8bf*)(qr + 32), *(const v8bf*)(qr + 40));
  }

  v8f oacc[2][4];
  float lsum[2][8];
#pragma unroll
  for (int rt = 0; rt < 2; ++rt) {
#pragma unroll
    for (int j = 0; j < 4; ++j) oacc[rt][j] = (v8f){0,0,0,0,0,0,0,0};
#pragma unroll
    for (int v = 0; v < 8; ++v) lsum[rt][v] = 0.0f;
  }

  __shared__ bf16 plds[32 * 32];
  const float sc = 0.125f / 50.0f;   // D^-0.5 / SOFTCLAMP

  for (int kt = 0; kt < NNT; kt += 32) {
    // prefetch next chunk: 32 K rows + 64 Vt row segments (1 cacheline each)
    if (kt + 32 < NNT) {
      __builtin_prefetch(kp + (long long)(kt + 32 + lane) * DD, 0, 3);
      __builtin_prefetch(vp + (long long)lane * NNT + kt + 32, 0, 3);
      __builtin_prefetch(vp + (long long)(lane + 32) * NNT + kt + 32, 0, 3);
    }

    // K B-fragments for key columns [kt..kt+15] and [kt+16..kt+31]
    const bf16* kc0 = kp + (long long)(kt +      l16) * DD + (half << 4);
    const bf16* kc1 = kp + (long long)(kt + 16 + l16) * DD + (half << 4);
    v16bf kb00 = *(const v16bf*)(kc0);
    v16bf kb01 = *(const v16bf*)(kc0 + 32);
    v16bf kb10 = *(const v16bf*)(kc1);
    v16bf kb11 = *(const v16bf*)(kc1 + 32);

#pragma unroll
    for (int rt = 0; rt < 2; ++rt) {
      v8f s0 = {0,0,0,0,0,0,0,0};
      v8f s1 = s0;
      s0 = wmma_bf16(aq[rt][0], kb00, s0);
      s0 = wmma_bf16(aq[rt][1], kb01, s0);
      s1 = wmma_bf16(aq[rt][0], kb10, s1);
      s1 = wmma_bf16(aq[rt][1], kb11, s1);
#pragma unroll
      for (int v = 0; v < 8; ++v) {
        float p0 = exp50(fast_tanh(s0[v] * sc));
        float p1 = exp50(fast_tanh(s1[v] * sc));
        lsum[rt][v] += p0 + p1;
        int row = rt * 16 + v + half * 8;
        plds[row * 32 + l16]      = (bf16)p0;
        plds[row * 32 + 16 + l16] = (bf16)p1;
      }
    }
    __syncthreads();   // single-wave workgroup: S_NOP; LDS in-order per wave

    // V B-fragments (shared by both row tiles), P as A-fragments from LDS
    const bf16* vbp = vp + (long long)l16 * NNT + kt + (half << 4);
    v16bf vb0 = *(const v16bf*)(vbp);
    v16bf vb1 = *(const v16bf*)(vbp + 16 * (long long)NNT);
    v16bf vb2 = *(const v16bf*)(vbp + 32 * (long long)NNT);
    v16bf vb3 = *(const v16bf*)(vbp + 48 * (long long)NNT);
#pragma unroll
    for (int rt = 0; rt < 2; ++rt) {
      const bf16* pr = &plds[(rt * 16 + l16) * 32 + (half << 4)];
      v16bf pa = cat8(*(const v8bf*)(pr), *(const v8bf*)(pr + 8));
      oacc[rt][0] = wmma_bf16(pa, vb0, oacc[rt][0]);
      oacc[rt][1] = wmma_bf16(pa, vb1, oacc[rt][1]);
      oacc[rt][2] = wmma_bf16(pa, vb2, oacc[rt][2]);
      oacc[rt][3] = wmma_bf16(pa, vb3, oacc[rt][3]);
    }
    __syncthreads();
  }

  // epilogue: one shuffle-reduction of the row sums, then O = acc / l
#pragma unroll
  for (int rt = 0; rt < 2; ++rt) {
    bf16* orow = O + (long long)(b * NNT + qt + rt * 16) * DI + h * DD;
#pragma unroll
    for (int v = 0; v < 8; ++v) {
      float ls = lsum[rt][v];
      ls += __shfl_xor(ls, 1, 32);
      ls += __shfl_xor(ls, 2, 32);
      ls += __shfl_xor(ls, 4, 32);
      ls += __shfl_xor(ls, 8, 32);
      float invl = 1.0f / ls;
      int row = v + half * 8;
      long long off = (long long)row * DI + l16;
      orow[off]      = (bf16)(oacc[rt][0][v] * invl);
      orow[off + 16] = (bf16)(oacc[rt][1][v] * invl);
      orow[off + 32] = (bf16)(oacc[rt][2][v] * invl);
      orow[off + 48] = (bf16)(oacc[rt][3][v] * invl);
    }
  }
}

// ---------------------------------------------------------------------------
extern "C" void kernel_launch(void* const* d_in, const int* in_sizes, int n_in,
                              void* d_out, int out_size, void* d_ws, size_t ws_size,
                              hipStream_t stream) {
  (void)in_sizes; (void)n_in; (void)out_size; (void)ws_size;
  const float* x0    = (const float*)d_in[0];
  const float* x1    = (const float*)d_in[1];
  const float* Wqkv0 = (const float*)d_in[2];
  const float* Wqkv1 = (const float*)d_in[3];
  const float* qg0   = (const float*)d_in[4];
  const float* kg0   = (const float*)d_in[5];
  const float* qg1   = (const float*)d_in[6];
  const float* kg1   = (const float*)d_in[7];
  const float* Wout0 = (const float*)d_in[8];
  const float* Wout1 = (const float*)d_in[9];

  // Deterministic workspace carve-up (bf16 elements), 256B aligned
  char* w = (char*)d_ws;
  size_t off = 0;
  auto alloc = [&](size_t elems) -> bf16* {
    bf16* p = (bf16*)(w + off);
    off += (elems * sizeof(bf16) + 255) & ~(size_t)255;
    return p;
  };
  bf16* x0b  = alloc((size_t)BB * NN0 * DIM0);
  bf16* x1b  = alloc((size_t)BB * NN1 * DIM1);
  bf16* wq0t = alloc((size_t)K3 * DIM0);
  bf16* wq1t = alloc((size_t)K3 * DIM1);
  bf16* wo0t = alloc((size_t)DIM0 * DI);
  bf16* wo1t = alloc((size_t)DIM1 * DI);
  bf16* qkv  = alloc((size_t)BB * NNT * K3);
  bf16* qbuf = alloc((size_t)BB * HH * NNT * DD);
  bf16* kbuf = alloc((size_t)BB * HH * NNT * DD);
  bf16* vtb  = alloc((size_t)BB * HH * DD * NNT);
  bf16* obuf = alloc((size_t)BB * NNT * DI);

  // 1) convert inputs / transpose weights to bf16
  cast_f32_bf16<<<2048, 256, 0, stream>>>(x0, x0b, (long long)BB * NN0 * DIM0);
  cast_f32_bf16<<<2048, 256, 0, stream>>>(x1, x1b, (long long)BB * NN1 * DIM1);
  transpose_f32_bf16<<<2048, 256, 0, stream>>>(Wqkv0, wq0t, DIM0, K3);
  transpose_f32_bf16<<<2048, 256, 0, stream>>>(Wqkv1, wq1t, DIM1, K3);
  transpose_f32_bf16<<<2048, 256, 0, stream>>>(Wout0, wo0t, DI, DIM0);
  transpose_f32_bf16<<<2048, 256, 0, stream>>>(Wout1, wo1t, DI, DIM1);

  // 2) QKV projections (write packed [b][2048][3072] bf16)
  gemm16x64<false><<<dim3(K3 / 64, NN0 / 16, BB), 32, 0, stream>>>(
      x0b, wq0t, qkv, NN0, K3, DIM0,
      (long long)NN0 * DIM0, (long long)NNT * K3);
  gemm16x64<false><<<dim3(K3 / 64, NN1 / 16, BB), 32, 0, stream>>>(
      x1b, wq1t, qkv + (size_t)NN0 * K3, NN1, K3, DIM1,
      (long long)NN1 * DIM1, (long long)NNT * K3);

  // 3) QK RMSNorm + split heads (+ V transpose)
  qknorm_split<<<(BB * NNT * HH) / 8, 256, 0, stream>>>(
      qkv, qg0, kg0, qg1, kg1, qbuf, kbuf, vtb);

  // 4) flash attention with tanh soft-clamp (fixed-offset softmax)
  attn_fwd<<<dim3(NNT / 32, HH, BB), 32, 0, stream>>>(qbuf, kbuf, vtb, obuf);

  // 5) output projections straight into d_out (f32, fully overwrites it)
  float* out0 = (float*)d_out;
  float* out1 = out0 + (size_t)BB * NN0 * DIM0;
  gemm16x64<true><<<dim3(DIM0 / 64, NN0 / 16, BB), 32, 0, stream>>>(
      obuf, wo0t, out0, NN0, DIM0, DI,
      (long long)NNT * DI, (long long)NN0 * DIM0);
  gemm16x64<true><<<dim3(DIM1 / 64, NN1 / 16, BB), 32, 0, stream>>>(
      obuf + (size_t)NN0 * DI, wo1t, out1, NN1, DIM1, DI,
      (long long)NNT * DI, (long long)NN1 * DIM1);
}
